// MultimodalGNN_79628693667937
// MI455X (gfx1250) — compile-verified, matrix operand order checked by reading, not run on hardware
//
#include <hip/hip_runtime.h>
#include <hip/hip_bf16.h>

typedef __attribute__((ext_vector_type(16))) __bf16 v16bf;
typedef __attribute__((ext_vector_type(8)))  float  v8f;

#define HW      16384          /* 128*128 */
#define NIMG    16             /* b*t */
#define CIN     128            /* 2*c */
#define COUT    64
#define PIX     (NIMG * HW)    /* 262144 pixels */
#define BN_EPS  1e-5f

/* workspace byte offsets */
#define WS_XBF   ((size_t)0)
#define WS_Y     ((size_t)PIX * CIN * 2)            /* 67,108,864 */
#define WS_STATS (WS_Y + (size_t)PIX * COUT * 4)    /* +67,108,864 */
#define WS_SS    (WS_STATS + 512)

__global__ void k_zero_stats(float* __restrict__ g) {
    g[threadIdx.x] = 0.0f;     /* 128 threads: sum[64], sumsq[64] */
}

/* Fused concat + f32->bf16 + transpose to pixel-major [PIX][128].
   Block: 256 threads, 32 pixels (all share one image n). */
__global__ void k_pack(const float* __restrict__ v, const float* __restrict__ m,
                       __bf16* __restrict__ xbf) {
    __shared__ __bf16 tile[32][132];            /* pad: p-stride 66 dwords, conflict-free */
    const int tid = threadIdx.x;
    const int p0  = blockIdx.x * 32;
    const int n   = p0 >> 14;
    const int hw0 = p0 & (HW - 1);
    const int pl  = tid & 31;
    const int cb  = tid >> 5;                   /* 0..7 */
    const float* vb = v + (size_t)n * COUT * HW + hw0 + pl;
    const float* mb = m + (size_t)n * COUT * HW + hw0 + pl;
#pragma unroll
    for (int i = 0; i < 8; ++i) {
        int c = cb + i * 8;                      /* 0..63 */
        tile[pl][c]      = (__bf16)vb[(size_t)c * HW];
        tile[pl][64 + c] = (__bf16)mb[(size_t)c * HW];
    }
    __syncthreads();
    unsigned* dst = (unsigned*)(xbf + (size_t)p0 * CIN);
#pragma unroll
    for (int i = 0; i < 8; ++i) {
        int d = tid + i * 256;                   /* dword id 0..2047 */
        int p = d >> 6, j = d & 63;
        dst[p * 64 + j] = *(const unsigned*)&tile[p][2 * j];
    }
}

/* GEMM: y[p][o] = sum_k Xbf[p][k]*W[o][k] + bias[o]; plus per-channel sum/sumsq.
   Block = 256 threads = 8 waves; each wave: 32 pixels (2 M-tiles) x 64 out-ch (4 N-tiles).
   32 wmma per wave; acc = 64 VGPRs -> no spills under the 128-VGPR budget. */
__global__ void k_gemm(const __bf16* __restrict__ xbf,
                       const float* __restrict__ w,      /* [64][128] */
                       const float* __restrict__ bias,   /* [64] */
                       float* __restrict__ y,            /* [PIX][64] */
                       float* __restrict__ gstats) {     /* sum[64], sumsq[64] */
    __shared__ __bf16 wlds[COUT][136];          /* bf16 weights, padded rows */
    __shared__ float  ssum[COUT], ssq[COUT];
    const int tid = threadIdx.x;
    if (tid < COUT) { ssum[tid] = 0.0f; ssq[tid] = 0.0f; }
#pragma unroll
    for (int i = 0; i < 32; ++i) {
        int idx = tid + i * 256;                 /* 0..8191 */
        wlds[idx >> 7][idx & 127] = (__bf16)w[idx];
    }
    __syncthreads();

    const int wave = tid >> 5, lane = tid & 31;
    const int p0   = blockIdx.x * 256 + wave * 32;
    const int lmod = lane & 15, lhi = lane >> 4;

    v8f acc[2][4];
#pragma unroll
    for (int mt = 0; mt < 2; ++mt)
#pragma unroll
        for (int nt = 0; nt < 4; ++nt)
            acc[mt][nt] = (v8f){0.f,0.f,0.f,0.f,0.f,0.f,0.f,0.f};

    /* A fragment per lane: lane<16 -> K 0-7 & 16-23 of the kc chunk; lane>=16 -> K 8-15 & 24-31 */
    const char* abase = (const char*)(xbf + (size_t)(p0 + lmod) * CIN) + lhi * 16;
#pragma unroll
    for (int kc = 0; kc < 4; ++kc) {
        v16bf a[2];
#pragma unroll
        for (int mt = 0; mt < 2; ++mt) {
            const char* arow = abase + (size_t)mt * 16 * CIN * 2 + kc * 64;
            ((uint4*)&a[mt])[0] = *(const uint4*)(arow);
            ((uint4*)&a[mt])[1] = *(const uint4*)(arow + 32);
        }
#pragma unroll
        for (int nt = 0; nt < 4; ++nt) {
            v16bf b;
            const __bf16* bp = &wlds[nt * 16 + lmod][kc * 32 + lhi * 8];
            ((uint4*)&b)[0] = *(const uint4*)bp;
            ((uint4*)&b)[1] = *(const uint4*)(bp + 16);
#pragma unroll
            for (int mt = 0; mt < 2; ++mt)
                acc[mt][nt] = __builtin_amdgcn_wmma_f32_16x16x32_bf16(
                    false, a[mt], false, b, (short)0, acc[mt][nt], false, false);
        }
    }

    /* epilogue: bias, store y, partial per-channel stats */
#pragma unroll
    for (int nt = 0; nt < 4; ++nt) {
        int o = nt * 16 + lmod;
        float bv = bias[o];
        float s = 0.0f, q = 0.0f;
#pragma unroll
        for (int mt = 0; mt < 2; ++mt) {
            float* yrow = y + (size_t)(p0 + mt * 16 + lhi * 8) * COUT + o;
#pragma unroll
            for (int r = 0; r < 8; ++r) {        /* C layout: VGPR r -> M = r + lhi*8 */
                float val = acc[mt][nt][r] + bv;
                yrow[(size_t)r * COUT] = val;
                s += val; q += val * val;
            }
        }
        atomicAdd(&ssum[o], s);
        atomicAdd(&ssq[o], q);
    }
    __syncthreads();
    if (tid < COUT) {
        atomicAdd(&gstats[tid],        ssum[tid]);
        atomicAdd(&gstats[COUT + tid], ssq[tid]);
    }
}

__global__ void k_stats(const float* __restrict__ gstats,
                        const float* __restrict__ gamma,
                        const float* __restrict__ beta,
                        float* __restrict__ ss) {        /* scale[64], shift[64] */
    int tid = threadIdx.x;
    if (tid < COUT) {
        const float invP = 1.0f / (float)PIX;
        float mean = gstats[tid] * invP;
        float var  = gstats[COUT + tid] * invP - mean * mean;
        float inv  = rsqrtf(var + BN_EPS);
        float sc   = inv * gamma[tid];
        ss[tid]        = sc;
        ss[COUT + tid] = beta[tid] - mean * sc;
    }
}

/* normalize + ReLU + transpose back to [N, C, H, W]; y mostly resident in L2 */
__global__ void k_out(const float* __restrict__ y,
                      const float* __restrict__ ss,
                      float* __restrict__ out) {
    __shared__ float tile[32][65];
    const int tid = threadIdx.x;
    const int p0  = blockIdx.x * 32;
    const int n   = p0 >> 14;
    const int hw0 = p0 & (HW - 1);
#pragma unroll
    for (int i = 0; i < 8; ++i) {
        int d = tid + i * 256;
        int p = d >> 6, o = d & 63;
        float val = y[(size_t)(p0 + p) * COUT + o];
        val = fmaxf(val * ss[o] + ss[COUT + o], 0.0f);
        tile[p][o] = val;
    }
    __syncthreads();
    const int pl = tid & 31, cb = tid >> 5;
    float* ob = out + (size_t)n * COUT * HW + hw0 + pl;
#pragma unroll
    for (int i = 0; i < 8; ++i) {
        int c = cb + i * 8;
        ob[(size_t)c * HW] = tile[pl][c];
    }
}

extern "C" void kernel_launch(void* const* d_in, const int* in_sizes, int n_in,
                              void* d_out, int out_size, void* d_ws, size_t ws_size,
                              hipStream_t stream) {
    const float* vpos  = (const float*)d_in[0];
    const float* mfeat = (const float*)d_in[1];
    const float* cw    = (const float*)d_in[2];
    const float* cbias = (const float*)d_in[3];
    const float* gamma = (const float*)d_in[4];
    const float* beta  = (const float*)d_in[5];
    float* out = (float*)d_out;

    char* ws = (char*)d_ws;
    __bf16* xbf   = (__bf16*)(ws + WS_XBF);
    float*  y     = (float*)(ws + WS_Y);
    float*  gst   = (float*)(ws + WS_STATS);
    float*  ssbuf = (float*)(ws + WS_SS);

    k_zero_stats<<<1, 128, 0, stream>>>(gst);
    k_pack<<<PIX / 32, 256, 0, stream>>>(vpos, mfeat, xbf);
    k_gemm<<<PIX / 256, 256, 0, stream>>>(xbf, cw, cbias, y, gst);
    k_stats<<<1, 64, 0, stream>>>(gst, gamma, beta, ssbuf);
    k_out<<<PIX / 32, 256, 0, stream>>>(y, ssbuf, out);
}
